// Generator_41953240547421
// MI455X (gfx1250) — compile-verified, matrix operand order checked by reading, not run on hardware
//
#include <hip/hip_runtime.h>

// ---------------------------------------------------------------------------
// Types for WMMA fragments (wave32, V_WMMA_F32_16X16X32_F16)
// ---------------------------------------------------------------------------
typedef __attribute__((ext_vector_type(16))) _Float16 v16h;
typedef __attribute__((ext_vector_type(8)))  _Float16 half8;
typedef __attribute__((ext_vector_type(8)))  float    v8f;

// Async global->LDS 16B copy (ASYNCcnt-tracked DMA, overlaps WMMA).
__device__ __forceinline__ void async_b128(const void* gptr, void* lptr) {
    unsigned lds = (unsigned)(unsigned long long)lptr;           // low 32b = LDS offset
    unsigned long long g = (unsigned long long)gptr;
    asm volatile("global_load_async_to_lds_b128 %0, %1, off"
                 :: "v"(lds), "v"(g) : "memory");
}
__device__ __forceinline__ void async_wait0() {
    asm volatile("s_wait_asynccnt 0x0" ::: "memory");
}

// Build a 16-half fragment from a k-contiguous LDS row (two ds_load_b128).
__device__ __forceinline__ v16h frag_row(const _Float16* base, int hf) {
    half8 lo = *(const half8*)(base + hf * 8);
    half8 hi = *(const half8*)(base + 16 + hf * 8);
    return __builtin_shufflevector(lo, hi, 0,1,2,3,4,5,6,7,8,9,10,11,12,13,14,15);
}

// ---------------------------------------------------------------------------
// Weight packing: f32 conv weights -> f16 GEMM B-matrix, N-MAJOR [Npad x Kpad]
// so async DMA chunks are k-contiguous. K index = (ci, ky, kx).
// transposed=1: torch ConvTranspose2d layout (in,out,kh,kw) + spatial flip.
// ---------------------------------------------------------------------------
__global__ __launch_bounds__(256) void pack_weights(
    const float* __restrict__ w, _Float16* __restrict__ bp,
    int Cout, int Cin, int kk, int K, int Kpad, int Npad, int transposed)
{
    int idx = blockIdx.x * 256 + threadIdx.x;
    int total = Kpad * Npad;
    if (idx >= total) return;
    int k = idx / Npad;
    int n = idx - k * Npad;
    float v = 0.f;
    if (k < K && n < Cout) {
        int kk2 = kk * kk;
        int ci = k / kk2;
        int r  = k - ci * kk2;
        int ky = r / kk;
        int kx = r - ky * kk;
        if (transposed)
            v = w[(((size_t)ci * Cout + n) * kk + (kk - 1 - ky)) * kk + (kk - 1 - kx)];
        else
            v = w[(((size_t)n * Cin + ci) * kk + ky) * kk + kx];
    }
    bp[(size_t)n * Kpad + k] = (_Float16)v;
}

// ---------------------------------------------------------------------------
// Implicit-GEMM conv, WMMA path. Block tile 64(M) x 128(N), 8 waves in a
// 2x4 grid, each wave owns a 2x2 set of 16x16 accumulators (4 WMMA/K-step,
// 2x fragment reuse). Double-buffered LDS, one barrier per K-step.
//   A tile (64x32, [m][k], f32->f16 im2col, VALU staged)
//   B tile (128x32, [n][k], async global->LDS DMA overlapped with WMMA)
// All fragments are two ds_load_b128 from k-contiguous LDS rows.
// dil==2 implements lhs_dilation=2 (transposed conv as dilated-input conv).
// ---------------------------------------------------------------------------
__global__ __launch_bounds__(256) void conv_wmma(
    const float* __restrict__ x, const _Float16* __restrict__ bp,
    float* __restrict__ y,
    int Cin, int Hin, int Win, int Cout, int Hout, int Wout,
    int kk, int stride, int pad, int dil, int K, int Kpad, int Npad)
{
    __shared__ __align__(16) _Float16 sA[2][64 * 40];    // [m][k]
    __shared__ __align__(16) _Float16 sB[2][128 * 40];   // [n][k]

    const int tid  = threadIdx.x;
    const int wave = tid >> 5, lane = tid & 31;
    const int wr   = wave >> 2, wc = wave & 3;           // 2 x 4 wave grid
    const int hf   = lane >> 4, l16 = lane & 15;
    const int M    = Hout * Wout;
    const int m0   = blockIdx.x * 64;
    const int n0   = blockIdx.y * 128;
    const int kk2  = kk * kk;

    v8f acc[2][2] = {};

    auto stageA = [&](int buf, int k0) {
        #pragma unroll
        for (int j = 0; j < 8; ++j) {                    // 64x32 elems, 8/thread
            int e  = tid * 8 + j;
            int ml = e >> 5, kl = e & 31;
            int kg = k0 + kl;
            float v = 0.f;
            if (kg < K) {
                int m  = m0 + ml;                        // M % 64 == 0 here
                int oy = m / Wout, ox = m - oy * Wout;
                int ci = kg / kk2;
                int r  = kg - ci * kk2;
                int ky = r / kk, kx = r - ky * kk;
                int ty = oy * stride - pad + ky;
                int tx = ox * stride - pad + kx;
                int iy, ix; bool ok;
                if (dil == 2) { ok = !((ty | tx) & 1); iy = ty >> 1; ix = tx >> 1; }
                else          { ok = true;             iy = ty;      ix = tx;      }
                ok = ok && iy >= 0 && iy < Hin && ix >= 0 && ix < Win;
                if (ok) v = x[((size_t)ci * Hin + iy) * Win + ix];
            }
            sA[buf][ml * 40 + kl] = (_Float16)v;
        }
    };
    auto stageB = [&](int buf, int k0) {
        #pragma unroll
        for (int j = 0; j < 2; ++j) {                    // 128 rows x 4 chunks
            int e  = tid * 2 + j;
            int nl = e >> 2, k8 = e & 3;                 // row, 16B chunk in row
            _Float16* dst = &sB[buf][nl * 40 + k8 * 8];
            if (n0 + nl < Npad)
                async_b128(bp + (size_t)(n0 + nl) * Kpad + k0 + k8 * 8, dst);
            else {
                half8 z = {};
                *(half8*)dst = z;
            }
        }
    };
    auto compute = [&](int buf) {
        v16h af[2], bf[2];
        #pragma unroll
        for (int ti = 0; ti < 2; ++ti)
            af[ti] = frag_row(&sA[buf][(wr * 32 + ti * 16 + l16) * 40], hf);
        #pragma unroll
        for (int tj = 0; tj < 2; ++tj)
            bf[tj] = frag_row(&sB[buf][(wc * 32 + tj * 16 + l16) * 40], hf);
        #pragma unroll
        for (int ti = 0; ti < 2; ++ti)
            #pragma unroll
            for (int tj = 0; tj < 2; ++tj)
                acc[ti][tj] = __builtin_amdgcn_wmma_f32_16x16x32_f16(
                    false, af[ti], false, bf[tj], (short)0, acc[ti][tj],
                    false, false);
    };

    const int nsteps = Kpad >> 5;
    stageB(0, 0);
    stageA(0, 0);
    async_wait0();
    __syncthreads();
    for (int s = 0; s < nsteps; ++s) {
        int cur = s & 1;
        if (s + 1 < nsteps) {                            // DMA next tile during WMMA
            stageB(cur ^ 1, (s + 1) * 32);
            stageA(cur ^ 1, (s + 1) * 32);
        }
        compute(cur);
        async_wait0();
        __syncthreads();
    }

    #pragma unroll
    for (int ti = 0; ti < 2; ++ti)
        #pragma unroll
        for (int tj = 0; tj < 2; ++tj) {
            int co = n0 + wc * 32 + tj * 16 + l16;
            if (co < Cout) {
                int mb = m0 + wr * 32 + ti * 16 + hf * 8;
                float* yp = y + (size_t)co * M + mb;
                #pragma unroll
                for (int r = 0; r < 8; ++r) yp[r] = acc[ti][tj][r];
            }
        }
}

// ---------------------------------------------------------------------------
// InstanceNorm (+ optional affine, ReLU, residual-add). One block per channel.
// ---------------------------------------------------------------------------
__global__ __launch_bounds__(256) void inorm_kernel(
    const float* __restrict__ x, float* __restrict__ y,
    const float* __restrict__ res,
    const float* __restrict__ gamma, const float* __restrict__ beta,
    int HW, int relu)
{
    __shared__ float r1[256], r2[256];
    const int c = blockIdx.x, tid = threadIdx.x;
    const float* xc = x + (size_t)c * HW;
    float s = 0.f, s2 = 0.f;
    for (int i = tid; i < HW; i += 256) { float v = xc[i]; s += v; s2 += v * v; }
    r1[tid] = s; r2[tid] = s2; __syncthreads();
    for (int st = 128; st > 0; st >>= 1) {
        if (tid < st) { r1[tid] += r1[tid + st]; r2[tid] += r2[tid + st]; }
        __syncthreads();
    }
    float mu  = r1[0] / HW;
    float var = r2[0] / HW - mu * mu;
    float rsg = rsqrtf(var + 1e-5f);
    float g = gamma ? gamma[c] : 1.f;
    float b = beta  ? beta[c]  : 0.f;
    for (int i = tid; i < HW; i += 256) {
        float v = (xc[i] - mu) * rsg * g + b;
        if (relu) v = fmaxf(v, 0.f);
        if (res)  v += res[(size_t)c * HW + i];
        y[(size_t)c * HW + i] = v;
    }
}

// ---------------------------------------------------------------------------
// SPADE 1x1 convs (256 -> 1), both maps in one pass.
// ---------------------------------------------------------------------------
__global__ __launch_bounds__(256) void spade_1x1(
    const float* __restrict__ sp, const float* __restrict__ wg,
    const float* __restrict__ wb, float* __restrict__ gma,
    float* __restrict__ bta, int C, int HW)
{
    int i = blockIdx.x * 256 + threadIdx.x;
    if (i >= HW) return;
    float a = 0.f, b = 0.f;
    for (int c = 0; c < C; ++c) {
        float v = sp[(size_t)c * HW + i];
        a += v * wg[c];
        b += v * wb[c];
    }
    gma[i] = a; bta[i] = b;
}

// ---------------------------------------------------------------------------
// Build packed f16 theta/phi, PIXEL-MAJOR [b][m][416] so async DMA chunks
// are k-contiguous.  k<256: feat*mask_ds*0.01 ; 256<=k<392: diff ; else 0.
// ---------------------------------------------------------------------------
__global__ __launch_bounds__(256) void pack_theta(
    const float* __restrict__ feat, const float* __restrict__ mask,
    const float* __restrict__ diff, _Float16* __restrict__ out)
{
    int idx = blockIdx.x * 256 + threadIdx.x;
    const int total = 3 * 416 * 4096;
    if (idx >= total) return;
    int b = idx / (416 * 4096);
    int r = idx - b * (416 * 4096);
    int k = r >> 12;
    int m = r & 4095;
    float v = 0.f;
    if (k < 256) {
        int yy = m >> 6, xx = m & 63;
        float mk = mask[(size_t)b * 65536 + (yy * 4) * 256 + xx * 4];
        v = feat[(size_t)k * 4096 + m] * mk * 0.01f;
    } else if (k < 392) {
        v = diff[((size_t)b * 136 + (k - 256)) * 4096 + m];
    }
    out[((size_t)b * 4096 + m) * 416 + k] = (_Float16)v;
}

__global__ __launch_bounds__(256) void gms_bts_kernel(
    const float* __restrict__ gma, const float* __restrict__ bta,
    const float* __restrict__ mask_s, float* __restrict__ gms,
    float* __restrict__ bts)
{
    int idx = blockIdx.x * 256 + threadIdx.x;
    if (idx >= 3 * 4096) return;
    int b = idx >> 12, m = idx & 4095;
    int yy = m >> 6, xx = m & 63;
    float mk = mask_s[(size_t)b * 65536 + (yy * 4) * 256 + xx * 4];
    gms[idx] = gma[m] * mk;
    bts[idx] = bta[m] * mk;
}

__global__ __launch_bounds__(256) void zero_kernel(float* __restrict__ p, int n)
{
    int i = blockIdx.x * 256 + threadIdx.x;
    if (i < n) p[i] = 0.f;
}

// ---------------------------------------------------------------------------
// Flash-style fused attention: w = theta^T phi (M=N=4096, K=416), online
// softmax(200*w) with dual numerator accumulation (gamma against gms, beta
// against bts). Never materializes the 3x4096x4096 matrix (201 MB > L2).
// Both tiles async-DMA'd global->LDS (row-major [idx][k]), double buffered.
// ---------------------------------------------------------------------------
__global__ __launch_bounds__(256) void attn_flash(
    const _Float16* __restrict__ theta, const _Float16* __restrict__ phi,
    const float* __restrict__ gms, const float* __restrict__ bts,
    float* __restrict__ gammaM, float* __restrict__ betaM)
{
    __shared__ __align__(16) _Float16 sA[2][32 * 40];    // [m][k]
    __shared__ __align__(16) _Float16 sB[2][64 * 40];    // [n][k]
    __shared__ float sW[32 * 72];
    __shared__ float sMx[256], sDen[256], sNg[256], sNb[256];

    const int b   = blockIdx.y;
    const int m0  = blockIdx.x * 32;
    const int tid = threadIdx.x;
    const int wave = tid >> 5, lane = tid & 31;
    const int wr = wave & 1, wc = wave >> 1;
    const int hf = lane >> 4, l16 = lane & 15;
    const int row = tid >> 3, grp = tid & 7;

    const _Float16* th = theta + (size_t)b * 4096 * 416;
    const _Float16* ph = phi   + (size_t)b * 4096 * 416;

    auto stage = [&](int buf, int ks, int n0c) {
        if (tid < 128) {                                 // theta 32 rows x 4 chunks
            int ml = tid >> 2, k8 = tid & 3;
            async_b128(th + (size_t)(m0 + ml) * 416 + ks + k8 * 8,
                       &sA[buf][ml * 40 + k8 * 8]);
        }
        {                                                // phi 64 rows x 4 chunks
            int nl = tid >> 2, k8 = tid & 3;
            async_b128(ph + (size_t)(n0c + nl) * 416 + ks + k8 * 8,
                       &sB[buf][nl * 40 + k8 * 8]);
        }
    };

    float mx = -1e30f, den = 0.f, ng = 0.f, nb = 0.f;

    for (int n0 = 0; n0 < 4096; n0 += 64) {
        v8f acc = {};
        stage(0, 0, n0);
        async_wait0();
        __syncthreads();
        for (int s = 0; s < 13; ++s) {                   // 416 / 32
            int cur = s & 1;
            if (s + 1 < 13) stage(cur ^ 1, (s + 1) * 32, n0);
            v16h av = frag_row(&sA[cur][(wr * 16 + l16) * 40], hf);
            v16h bv = frag_row(&sB[cur][(wc * 16 + l16) * 40], hf);
            acc = __builtin_amdgcn_wmma_f32_16x16x32_f16(false, av, false, bv,
                                                         (short)0, acc, false, false);
            async_wait0();
            __syncthreads();
        }
        // dump 32x64 score tile
        #pragma unroll
        for (int r = 0; r < 8; ++r)
            sW[(wr * 16 + hf * 8 + r) * 72 + wc * 16 + l16] = acc[r];
        __syncthreads();
        // online softmax + dual numerators, 8 cols per thread
        #pragma unroll
        for (int j = 0; j < 8; ++j) {
            int col = grp * 8 + j;
            float v = 200.f * sW[row * 72 + col];
            int n   = n0 + col;
            float g = gms[(size_t)b * 4096 + n];
            float t = bts[(size_t)b * 4096 + n];
            if (v > mx) {
                float sc = __expf(mx - v);
                den = den * sc + 1.f;
                ng  = ng  * sc + g;
                nb  = nb  * sc + t;
                mx  = v;
            } else {
                float e = __expf(v - mx);
                den += e; ng += e * g; nb += e * t;
            }
        }
        __syncthreads();
    }

    sMx[tid] = mx; sDen[tid] = den; sNg[tid] = ng; sNb[tid] = nb;
    __syncthreads();
    if (grp == 0) {
        float M = -1e30f;
        #pragma unroll
        for (int j = 0; j < 8; ++j) M = fmaxf(M, sMx[row * 8 + j]);
        float D = 0.f, G = 0.f, B = 0.f;
        #pragma unroll
        for (int j = 0; j < 8; ++j) {
            float sc = __expf(sMx[row * 8 + j] - M);
            D += sDen[row * 8 + j] * sc;
            G += sNg[row * 8 + j] * sc;
            B += sNb[row * 8 + j] * sc;
        }
        atomicAdd(&gammaM[m0 + row], G / D);
        atomicAdd(&betaM[m0 + row],  B / D);
    }
}

__global__ __launch_bounds__(256) void apply_attn(
    float* __restrict__ ct, const float* __restrict__ gammaM,
    const float* __restrict__ betaM)
{
    int idx = blockIdx.x * 256 + threadIdx.x;
    if (idx >= 256 * 4096) return;
    int m = idx & 4095;
    ct[idx] = ct[idx] * (1.f + gammaM[m]) + betaM[m];
}

__global__ __launch_bounds__(256) void tanh_kernel(
    const float* __restrict__ x, float* __restrict__ y, int n)
{
    int i = blockIdx.x * 256 + threadIdx.x;
    if (i < n) y[i] = tanhf(x[i]);
}

// ---------------------------------------------------------------------------
// Host orchestration (params flattened in setup_inputs dict-insertion order)
// ---------------------------------------------------------------------------
enum {
    IN_C = 0, IN_S, IN_MC, IN_MS, IN_DC, IN_DS,
    P_PIN_W = 6, P_PIN_G, P_PIN_B, P_TIN_W,
    P_PD0_W = 10, P_PD0_G, P_PD0_B, P_TD0_W,
    P_PD1_W = 14, P_PD1_G, P_PD1_B, P_TD1_W,
    P_SPG = 18, P_SPB = 19,
    P_PBN0 = 20,   // + 6*i : {w1,g1,b1,w2,g2,b2}
    P_TBN0 = 38,   // + 2*i : {w1,w2}
    P_UP0 = 50, P_UP1 = 51, P_OUTW = 52
};

extern "C" void kernel_launch(void* const* d_in, const int* in_sizes, int n_in,
                              void* d_out, int out_size, void* d_ws, size_t ws_size,
                              hipStream_t stream)
{
    (void)in_sizes; (void)n_in; (void)out_size; (void)ws_size;
    char* ws = (char*)d_ws;
    size_t off = 0;
    auto alloc = [&](size_t bytes) -> size_t {
        off = (off + 255) & ~(size_t)255;
        size_t o = off; off += bytes; return o;
    };

    // packed f16 weights (N-major [Npad x Kpad])
    const size_t W_TIN = alloc(160u * 64 * 2);
    const size_t W_PIN = alloc(160u * 64 * 2);
    const size_t W_PD0 = alloc(1024u * 128 * 2);
    const size_t W_TD0 = alloc(1024u * 128 * 2);
    const size_t W_PD1 = alloc(2048u * 256 * 2);
    const size_t W_TD1 = alloc(2048u * 256 * 2);
    size_t W_PBN[3][2], W_TBN[6][2];
    for (int i = 0; i < 3; ++i) for (int j = 0; j < 2; ++j) W_PBN[i][j] = alloc(2304u * 256 * 2);
    for (int i = 0; i < 6; ++i) for (int j = 0; j < 2; ++j) W_TBN[i][j] = alloc(2304u * 256 * 2);
    const size_t W_UP0 = alloc(4096u * 128 * 2);
    const size_t W_UP1 = alloc(2048u * 64 * 2);
    const size_t W_OUT = alloc(3136u * 16 * 2);

    // activations (f32) and attention buffers
    const size_t T256 = alloc(64u * 65536 * 4);
    const size_t P256 = alloc(64u * 65536 * 4);
    const size_t T128 = alloc(128u * 16384 * 4);
    const size_t P128 = alloc(128u * 16384 * 4);
    const size_t T64  = alloc(256u * 4096 * 4);
    const size_t P64  = alloc(256u * 4096 * 4);
    const size_t H1   = alloc(256u * 4096 * 4);
    const size_t H2   = alloc(256u * 4096 * 4);
    const size_t GMA  = alloc(4096u * 4);
    const size_t BTA  = alloc(4096u * 4);
    const size_t GMS  = alloc(3u * 4096 * 4);
    const size_t BTS  = alloc(3u * 4096 * 4);
    const size_t GAM  = alloc(4096u * 4);
    const size_t BET  = alloc(4096u * 4);
    const size_t THP  = alloc(3u * 416 * 4096 * 2);
    const size_t PHP  = alloc(3u * 416 * 4096 * 2);
    const size_t OUTB = alloc(3u * 65536 * 4);

    auto F  = [&](size_t o) -> float* { return (float*)(ws + o); };
    auto Hh = [&](size_t o) -> _Float16* { return (_Float16*)(ws + o); };
    auto IN = [&](int i) -> const float* { return (const float*)d_in[i]; };

    auto pack = [&](int inIdx, size_t wOff, int Cout, int Cin, int kk,
                    int K, int Kpad, int Npad, int tr) {
        int total = Kpad * Npad;
        pack_weights<<<(total + 255) / 256, 256, 0, stream>>>(
            IN(inIdx), Hh(wOff), Cout, Cin, kk, K, Kpad, Npad, tr);
    };
    auto conv = [&](const float* x, size_t wOff, float* y,
                    int Cin, int Hin, int Win, int Cout, int Hout, int Wout,
                    int kk, int st, int pd, int dl, int K, int Kpad, int Npad) {
        dim3 g((unsigned)((Hout * Wout) / 64), (unsigned)((Npad + 127) / 128));
        conv_wmma<<<g, 256, 0, stream>>>(x, Hh(wOff), y, Cin, Hin, Win,
                                         Cout, Hout, Wout, kk, st, pd, dl,
                                         K, Kpad, Npad);
    };
    auto inorm = [&](size_t xo, size_t yo, const float* res,
                     const float* g, const float* b, int C, int HW, int relu) {
        inorm_kernel<<<C, 256, 0, stream>>>(F(xo), F(yo), res, g, b, HW, relu);
    };

    // ---- pack all weights ----
    pack(P_TIN_W, W_TIN, 64, 3, 7, 147, 160, 64, 0);
    pack(P_PIN_W, W_PIN, 64, 3, 7, 147, 160, 64, 0);
    pack(P_PD0_W, W_PD0, 128, 64, 4, 1024, 1024, 128, 0);
    pack(P_TD0_W, W_TD0, 128, 64, 4, 1024, 1024, 128, 0);
    pack(P_PD1_W, W_PD1, 256, 128, 4, 2048, 2048, 256, 0);
    pack(P_TD1_W, W_TD1, 256, 128, 4, 2048, 2048, 256, 0);
    for (int i = 0; i < 3; ++i) {
        pack(P_PBN0 + 6 * i + 0, W_PBN[i][0], 256, 256, 3, 2304, 2304, 256, 0);
        pack(P_PBN0 + 6 * i + 3, W_PBN[i][1], 256, 256, 3, 2304, 2304, 256, 0);
    }
    for (int i = 0; i < 6; ++i) {
        pack(P_TBN0 + 2 * i + 0, W_TBN[i][0], 256, 256, 3, 2304, 2304, 256, 0);
        pack(P_TBN0 + 2 * i + 1, W_TBN[i][1], 256, 256, 3, 2304, 2304, 256, 0);
    }
    pack(P_UP0, W_UP0, 128, 256, 4, 4096, 4096, 128, 1);
    pack(P_UP1, W_UP1, 64, 128, 4, 2048, 2048, 64, 1);
    pack(P_OUTW, W_OUT, 3, 64, 7, 3136, 3136, 16, 0);

    // ---- stems: 7x7 pad3, 3 -> 64 @256 ----
    conv(IN(IN_C), W_TIN, F(T256), 3, 256, 256, 64, 256, 256, 7, 1, 3, 1, 147, 160, 64);
    inorm(T256, T256, nullptr, nullptr, nullptr, 64, 65536, 1);
    conv(IN(IN_S), W_PIN, F(P256), 3, 256, 256, 64, 256, 256, 7, 1, 3, 1, 147, 160, 64);
    inorm(P256, P256, nullptr, IN(P_PIN_G), IN(P_PIN_B), 64, 65536, 1);

    // ---- downsample x2: 4x4 s2 p1 ----
    conv(F(P256), W_PD0, F(P128), 64, 256, 256, 128, 128, 128, 4, 2, 1, 1, 1024, 1024, 128);
    inorm(P128, P128, nullptr, IN(P_PD0_G), IN(P_PD0_B), 128, 16384, 1);
    conv(F(T256), W_TD0, F(T128), 64, 256, 256, 128, 128, 128, 4, 2, 1, 1, 1024, 1024, 128);
    inorm(T128, T128, nullptr, nullptr, nullptr, 128, 16384, 1);
    conv(F(P128), W_PD1, F(P64), 128, 128, 128, 256, 64, 64, 4, 2, 1, 1, 2048, 2048, 256);
    inorm(P64, P64, nullptr, IN(P_PD1_G), IN(P_PD1_B), 256, 4096, 1);
    conv(F(T128), W_TD1, F(T64), 128, 128, 128, 256, 64, 64, 4, 2, 1, 1, 2048, 2048, 256);
    inorm(T64, T64, nullptr, nullptr, nullptr, 256, 4096, 1);

    auto resblock = [&](size_t X, size_t w1, size_t w2,
                        const float* g1, const float* b1,
                        const float* g2, const float* b2) {
        conv(F(X), w1, F(H1), 256, 64, 64, 256, 64, 64, 3, 1, 1, 1, 2304, 2304, 256);
        inorm(H1, H1, nullptr, g1, b1, 256, 4096, 1);
        conv(F(H1), w2, F(H2), 256, 64, 64, 256, 64, 64, 3, 1, 1, 1, 2304, 2304, 256);
        inorm(H2, X, F(X), g2, b2, 256, 4096, 0);   // y = x + inorm(conv2)
    };

    // ---- resblocks i = 0..2 (pnet affine, tnet plain) ----
    for (int i = 0; i < 3; ++i) {
        int p = P_PBN0 + 6 * i;
        resblock(P64, W_PBN[i][0], W_PBN[i][1], IN(p + 1), IN(p + 2), IN(p + 4), IN(p + 5));
        resblock(T64, W_TBN[i][0], W_TBN[i][1], nullptr, nullptr, nullptr, nullptr);
    }

    // ---- i == 3: SPADE + region attention ----
    spade_1x1<<<(4096 + 255) / 256, 256, 0, stream>>>(
        F(P64), IN(P_SPG), IN(P_SPB), F(GMA), F(BTA), 256, 4096);
    {
        int total = 3 * 416 * 4096;
        pack_theta<<<(total + 255) / 256, 256, 0, stream>>>(F(T64), IN(IN_MC), IN(IN_DC), Hh(THP));
        pack_theta<<<(total + 255) / 256, 256, 0, stream>>>(F(P64), IN(IN_MS), IN(IN_DS), Hh(PHP));
    }
    gms_bts_kernel<<<(3 * 4096 + 255) / 256, 256, 0, stream>>>(
        F(GMA), F(BTA), IN(IN_MS), F(GMS), F(BTS));
    zero_kernel<<<16, 256, 0, stream>>>(F(GAM), 4096);
    zero_kernel<<<16, 256, 0, stream>>>(F(BET), 4096);
    attn_flash<<<dim3(128, 3), 256, 0, stream>>>(
        Hh(THP), Hh(PHP), F(GMS), F(BTS), F(GAM), F(BET));
    apply_attn<<<(256 * 4096) / 256, 256, 0, stream>>>(F(T64), F(GAM), F(BET));

    // ---- resblocks i = 3..5 (tnet only) ----
    for (int i = 3; i < 6; ++i)
        resblock(T64, W_TBN[i][0], W_TBN[i][1], nullptr, nullptr, nullptr, nullptr);

    // ---- upsample x2: ConvTranspose2d(4,2,1) == dil-2-input conv, pad 2 ----
    conv(F(T64), W_UP0, F(T128), 256, 64, 64, 128, 128, 128, 4, 1, 2, 2, 4096, 4096, 128);
    inorm(T128, T128, nullptr, nullptr, nullptr, 128, 16384, 1);
    conv(F(T128), W_UP1, F(T256), 128, 128, 128, 64, 256, 256, 4, 1, 2, 2, 2048, 2048, 64);
    inorm(T256, T256, nullptr, nullptr, nullptr, 64, 65536, 1);

    // ---- output conv 7x7 (64 -> 3) + tanh ----
    conv(F(T256), W_OUT, F(OUTB), 64, 256, 256, 3, 256, 256, 7, 1, 3, 1, 3136, 3136, 16);
    tanh_kernel<<<(196608 + 255) / 256, 256, 0, stream>>>(F(OUTB), (float*)d_out, 196608);
}
